// DeltaNetBlock_79302276153504
// MI455X (gfx1250) — matrix operand order repaired and checked by reading, not compiled, not run on hardware
//
#include <hip/hip_runtime.h>
#include <hip/hip_bf16.h>
#include <cstdint>

typedef __attribute__((ext_vector_type(4)))  float   v4f;
typedef __attribute__((ext_vector_type(8)))  float   v8f;
typedef __attribute__((ext_vector_type(8)))  __bf16  v8bf;
typedef __attribute__((ext_vector_type(16))) __bf16  v16bf;

#define D_DIM 1024
#define B_DIM 8
#define T_DIM 1024
#define M_DIM (B_DIM * T_DIM)

__device__ __forceinline__ unsigned short f2bf(float f) {
    unsigned int u = __builtin_bit_cast(unsigned int, f);
    u += 0x7FFFu + ((u >> 16) & 1u);           // round-to-nearest-even
    return (unsigned short)(u >> 16);
}

// VALU cross-lane add via DPP (no DS pipe, no waits).
#define DPP_ADD_F(v, ctrl)                                                    \
    (v) += __builtin_bit_cast(float, __builtin_amdgcn_update_dpp(             \
               0, __builtin_bit_cast(int, (v)), (ctrl), 0xf, 0xf, true))

__device__ __forceinline__ float wred16(float v) {   // sum within 16-lane group
    DPP_ADD_F(v, 0xB1);    // quad_perm [1,0,3,2]  : xor 1
    DPP_ADD_F(v, 0x4E);    // quad_perm [2,3,0,1]  : xor 2
    DPP_ADD_F(v, 0x141);   // row_half_mirror      : xor 7  (== xor 4 here)
    DPP_ADD_F(v, 0x140);   // row_mirror           : xor 15 (== xor 8 here)
    return v;
}
__device__ __forceinline__ float wred32(float v) {   // full wave32 sum
    v = wred16(v);
    unsigned int i = __builtin_bit_cast(unsigned int, v);
    unsigned int x = __builtin_amdgcn_permlanex16(i, i, 0x76543210u, 0xFEDCBA98u,
                                                  false, false); // swap halves
    return v + __builtin_bit_cast(float, x);
}

// ---------------------------------------------------------------------------
// Kernel 1: fused 4-way projection GEMM via bf16 WMMA.
//   z=0..2: out = x@W + bias  (q,k,v)  -> f32 buffers
//   z=3   : per-block partial of relu(x@Wb1+bb1) . Wb2  (beta pre-activation)
// Block tile 64x64, BK=32, 4 waves (2x2), each wave 2x2 wmma tiles.
// ---------------------------------------------------------------------------
#define BM 64
#define BN 64
#define BK 32
#define KP 40   // padded LDS row stride in u16 (80B: conflict-free, 16B aligned)

__global__ __launch_bounds__(128)
void proj_gemm(const float* __restrict__ x,
               const float* __restrict__ Wq, const float* __restrict__ bq,
               const float* __restrict__ Wk, const float* __restrict__ bk,
               const float* __restrict__ Wv, const float* __restrict__ bv,
               const float* __restrict__ Wb1, const float* __restrict__ bb1,
               const float* __restrict__ Wb2,
               float* __restrict__ qbuf, float* __restrict__ kbuf,
               float* __restrict__ vbuf, float* __restrict__ beta_part)
{
    __shared__ unsigned short a_s[BM * KP];
    __shared__ unsigned short b_s[BN * KP];
    __shared__ float pb[BM][2];

    const int z = blockIdx.z;
    const float* W;  const float* bias;  float* out;
    if      (z == 0) { W = Wq;  bias = bq;  out = qbuf; }
    else if (z == 1) { W = Wk;  bias = bk;  out = kbuf; }
    else if (z == 2) { W = Wv;  bias = bv;  out = vbuf; }
    else             { W = Wb1; bias = bb1; out = nullptr; }

    const int m0   = blockIdx.x * BM;
    const int n0   = blockIdx.y * BN;
    const int tid  = threadIdx.x;
    const int lane = tid & 31;
    const int wave = tid >> 5;
    const int wm   = wave >> 1;          // wave row (0..1)
    const int wn   = wave & 1;           // wave col (0..1)
    const int hi16 = (lane >> 4) & 1;

    v8f acc[2][2] = {};

    for (int kb0 = 0; kb0 < D_DIM; kb0 += BK) {
        // ---- stage A tile (64x32 f32 -> bf16, row-major) ----
        {
            const int c4 = tid & 7;        // float4 column
            const int rr = tid >> 3;       // 0..15
            #pragma unroll
            for (int p = 0; p < 4; ++p) {
                const int row = rr + p * 16;
                const v4f f = *reinterpret_cast<const v4f*>(
                    x + (size_t)(m0 + row) * D_DIM + kb0 + c4 * 4);
                unsigned int lo = f2bf(f[0]) | ((unsigned int)f2bf(f[1]) << 16);
                unsigned int hi = f2bf(f[2]) | ((unsigned int)f2bf(f[3]) << 16);
                unsigned int* dst = reinterpret_cast<unsigned int*>(&a_s[row * KP + c4 * 4]);
                dst[0] = lo; dst[1] = hi;
            }
        }
        // ---- stage B tile transposed: W[k][n] -> b_s[n*KP + k] ----
        {
            const int n4 = tid & 15;       // float4 column among 16
            const int kk = tid >> 4;       // 0..7
            #pragma unroll
            for (int p = 0; p < 4; ++p) {
                const int k = kk + p * 8;
                const v4f f = *reinterpret_cast<const v4f*>(
                    W + (size_t)(kb0 + k) * D_DIM + n0 + n4 * 4);
                b_s[(n4 * 4 + 0) * KP + k] = f2bf(f[0]);
                b_s[(n4 * 4 + 1) * KP + k] = f2bf(f[1]);
                b_s[(n4 * 4 + 2) * KP + k] = f2bf(f[2]);
                b_s[(n4 * 4 + 3) * KP + k] = f2bf(f[3]);
            }
        }
        __syncthreads();

        // ---- build fragments per documented 16-bit layouts ----
        // A 16x32: lanes 0-15 row=lane hold K {0..7,16..23}; lanes 16-31 hold K {8..15,24..31}
        v16bf afrag[2];
        #pragma unroll
        for (int tm = 0; tm < 2; ++tm) {
            const int row = wm * 32 + tm * 16 + (lane & 15);
            const int kb2 = hi16 ? 8 : 0;
            v8bf lo = *reinterpret_cast<const v8bf*>(&a_s[row * KP + kb2]);
            v8bf hi = *reinterpret_cast<const v8bf*>(&a_s[row * KP + kb2 + 16]);
            #pragma unroll
            for (int i = 0; i < 8; ++i) { afrag[tm][i] = lo[i]; afrag[tm][8 + i] = hi[i]; }
        }
        // B 32x16: lanes 0-15 col=lane hold K {0..7,16..23}; lanes 16-31 hold K {8..15,24..31}
        v16bf bfrag[2];
        #pragma unroll
        for (int tn = 0; tn < 2; ++tn) {
            const int col  = wn * 32 + tn * 16 + (lane & 15);
            const int koff = hi16 ? 8 : 0;
            v8bf lo = *reinterpret_cast<const v8bf*>(&b_s[col * KP + koff]);
            v8bf hi = *reinterpret_cast<const v8bf*>(&b_s[col * KP + koff + 16]);
            #pragma unroll
            for (int i = 0; i < 8; ++i) { bfrag[tn][i] = lo[i]; bfrag[tn][8 + i] = hi[i]; }
        }

        #pragma unroll
        for (int tm = 0; tm < 2; ++tm)
            #pragma unroll
            for (int tn = 0; tn < 2; ++tn)
                acc[tm][tn] = __builtin_amdgcn_wmma_f32_16x16x32_bf16(
                    false, afrag[tm], false, bfrag[tn],
                    (short)0, acc[tm][tn], false, false);

        __syncthreads();
    }

    // ---- epilogue ----
    // C/D 16x16 f32: lanes 0-15 N=lane, M=r; lanes 16-31 N=lane-16, M=r+8
    if (z < 3) {
        #pragma unroll
        for (int tm = 0; tm < 2; ++tm)
            #pragma unroll
            for (int tn = 0; tn < 2; ++tn) {
                const int col  = n0 + wn * 32 + tn * 16 + (lane & 15);
                const float bc = bias[col];
                #pragma unroll
                for (int r = 0; r < 8; ++r) {
                    const int row = m0 + wm * 32 + tm * 16 + r + hi16 * 8;
                    out[(size_t)row * D_DIM + col] = acc[tm][tn][r] + bc;
                }
            }
    } else {
        // deterministic per-block beta partial: sum_n relu(h1[row,n]) * Wb2[n]
        #pragma unroll
        for (int tm = 0; tm < 2; ++tm)
            #pragma unroll
            for (int r = 0; r < 8; ++r) {
                float s = 0.f;
                #pragma unroll
                for (int tn = 0; tn < 2; ++tn) {
                    const int col = n0 + wn * 32 + tn * 16 + (lane & 15);
                    float h = acc[tm][tn][r] + bias[col];
                    h = h > 0.f ? h : 0.f;
                    s += h * Wb2[col];
                }
                s = wred16(s);   // sum over the 16 lanes of this C half
                const int lrow = wm * 32 + tm * 16 + r + hi16 * 8;
                if ((lane & 15) == 0) pb[lrow][wn] = s;
            }
        __syncthreads();
        if (tid < 64)
            beta_part[(size_t)(m0 + tid) * 16 + blockIdx.y] = pb[tid][0] + pb[tid][1];
    }
}

// ---------------------------------------------------------------------------
// Kernel 2: beta = sigmoid(sum(partials) + bb2)
// ---------------------------------------------------------------------------
__global__ __launch_bounds__(256)
void beta_finalize(const float* __restrict__ part, const float* __restrict__ bb2,
                   float* __restrict__ beta)
{
    const int i = blockIdx.x * 256 + threadIdx.x;
    float s = bb2[0];
    #pragma unroll
    for (int nb = 0; nb < 16; ++nb) s += part[(size_t)i * 16 + nb];
    beta[i] = 1.0f / (1.0f + __expf(-s));
}

// ---------------------------------------------------------------------------
// Kernel 3: sequential delta-rule scan, S register-resident, row-partitioned.
// grid = B*32 WGs of 256 threads; each wave owns 4 rows of S (striped 32/lane).
// Cross-lane reductions are pure-VALU (DPP + permlanex16).
// Loads are software-pipelined one step ahead (double-buffered k/q/v/beta).
// Writes delta_out into d_out[0 .. M*D) and S_final into d_out[M*D ..).
// ---------------------------------------------------------------------------
__global__ __launch_bounds__(256)
__attribute__((amdgpu_waves_per_eu(2)))
void delta_scan(const float* __restrict__ qb, const float* __restrict__ kb,
                const float* __restrict__ vb, const float* __restrict__ beta,
                float* __restrict__ dout)
{
    const int b    = blockIdx.x >> 5;
    const int rb   = blockIdx.x & 31;
    const int lane = threadIdx.x & 31;
    const int wave = threadIdx.x >> 5;
    const int row0 = rb * 32 + wave * 4;

    float S[4][32];
    #pragma unroll
    for (int r = 0; r < 4; ++r)
        #pragma unroll
        for (int j = 0; j < 32; ++j) S[r][j] = 0.f;

    const size_t baseBT = (size_t)b * T_DIM;
    float* out  = dout;
    float* sfin = dout + (size_t)M_DIM * D_DIM;

    // issue loads for step t into the given register buffers
    auto LOAD = [&](int t, float (&ks)[32], float (&qs)[32],
                    float (&vr)[4], float& bt) {
        const size_t off = (baseBT + t) * D_DIM;
        const v4f* kp = reinterpret_cast<const v4f*>(kb + off + lane * 32);
        const v4f* qp = reinterpret_cast<const v4f*>(qb + off + lane * 32);
        #pragma unroll
        for (int c = 0; c < 8; ++c) {
            const v4f kv = kp[c];
            const v4f qv = qp[c];
            #pragma unroll
            for (int e = 0; e < 4; ++e) { ks[c*4+e] = kv[e]; qs[c*4+e] = qv[e]; }
        }
        #pragma unroll
        for (int r = 0; r < 4; ++r) vr[r] = vb[off + row0 + r];
        bt = beta[baseBT + t];
    };

    // one recurrence step using pre-loaded operands
    auto STEP = [&](int t, const float (&ks)[32], const float (&qs)[32],
                    const float (&vr)[4], float bt) {
        float dk[4] = {0,0,0,0}, dq[4] = {0,0,0,0}, kq = 0.f;
        #pragma unroll
        for (int j = 0; j < 32; ++j) {
            kq += ks[j] * qs[j];
            #pragma unroll
            for (int r = 0; r < 4; ++r) {
                dk[r] += S[r][j] * ks[j];
                dq[r] += S[r][j] * qs[j];
            }
        }
        #pragma unroll
        for (int r = 0; r < 4; ++r) { dk[r] = wred32(dk[r]); dq[r] = wred32(dq[r]); }
        kq = wred32(kq);

        const size_t off = (baseBT + t) * D_DIM;
        #pragma unroll
        for (int r = 0; r < 4; ++r) {
            const float u = bt * (vr[r] - dk[r]);
            #pragma unroll
            for (int j = 0; j < 32; ++j) S[r][j] += u * ks[j];
            const float o = dq[r] + u * kq;   // o = S_new.q = S.q + u*(k.q)
            if (lane == r) out[off + row0 + r] = o;
        }
    };

    float k0[32], q0[32], v0r[4], b0;
    float k1[32], q1[32], v1r[4], b1;

    LOAD(0, k0, q0, v0r, b0);
    for (int t = 0; t < T_DIM; t += 2) {
        LOAD(t + 1, k1, q1, v1r, b1);          // prefetch next step
        STEP(t, k0, q0, v0r, b0);
        if (t + 2 < T_DIM) LOAD(t + 2, k0, q0, v0r, b0);
        STEP(t + 1, k1, q1, v1r, b1);
    }

    // S_final: written once, never re-read -> non-temporal stores
    #pragma unroll
    for (int r = 0; r < 4; ++r) {
        v4f* p = reinterpret_cast<v4f*>(
            sfin + ((size_t)b * D_DIM + (row0 + r)) * D_DIM + lane * 32);
        #pragma unroll
        for (int c = 0; c < 8; ++c) {
            v4f v4;
            v4[0] = S[r][c*4+0]; v4[1] = S[r][c*4+1];
            v4[2] = S[r][c*4+2]; v4[3] = S[r][c*4+3];
            __builtin_nontemporal_store(v4, p + c);
        }
    }
}

// ---------------------------------------------------------------------------
// Kernel 4: in-place LayerNorm over d_out[0 .. M*D), one WG per row.
// ---------------------------------------------------------------------------
__global__ __launch_bounds__(256)
void layernorm_inplace(float* __restrict__ y, const float* __restrict__ w,
                       const float* __restrict__ bvec)
{
    __shared__ float red[8];
    const int tid = threadIdx.x;
    float* p = y + (size_t)blockIdx.x * D_DIM;

    const v4f v = reinterpret_cast<const v4f*>(p)[tid];
    float s = wred32(v[0] + v[1] + v[2] + v[3]);
    if ((tid & 31) == 0) red[tid >> 5] = s;
    __syncthreads();
    float mu = 0.f;
    #pragma unroll
    for (int i = 0; i < 8; ++i) mu += red[i];
    mu *= (1.0f / D_DIM);

    const float dx = v[0] - mu, dy = v[1] - mu, dz = v[2] - mu, dw = v[3] - mu;
    float ss = wred32(dx*dx + dy*dy + dz*dz + dw*dw);
    __syncthreads();
    if ((tid & 31) == 0) red[tid >> 5] = ss;
    __syncthreads();
    float var = 0.f;
    #pragma unroll
    for (int i = 0; i < 8; ++i) var += red[i];
    var *= (1.0f / D_DIM);
    const float inv = rsqrtf(var + 1e-5f);

    const v4f wv = reinterpret_cast<const v4f*>(w)[tid];
    const v4f bv = reinterpret_cast<const v4f*>(bvec)[tid];
    v4f o;
    o[0] = dx * inv * wv[0] + bv[0];
    o[1] = dy * inv * wv[1] + bv[1];
    o[2] = dz * inv * wv[2] + bv[2];
    o[3] = dw * inv * wv[3] + bv[3];
    reinterpret_cast<v4f*>(p)[tid] = o;
}

// ---------------------------------------------------------------------------
extern "C" void kernel_launch(void* const* d_in, const int* in_sizes, int n_in,
                              void* d_out, int out_size, void* d_ws, size_t ws_size,
                              hipStream_t stream)
{
    const float* x   = (const float*)d_in[0];
    const float* Wq  = (const float*)d_in[1];
    const float* bq  = (const float*)d_in[2];
    const float* Wk  = (const float*)d_in[3];
    const float* bk  = (const float*)d_in[4];
    const float* Wv  = (const float*)d_in[5];
    const float* bv  = (const float*)d_in[6];
    const float* Wb1 = (const float*)d_in[7];
    const float* bb1 = (const float*)d_in[8];
    const float* Wb2 = (const float*)d_in[9];
    const float* bb2 = (const float*)d_in[10];
    const float* lnw = (const float*)d_in[11];
    const float* lnb = (const float*)d_in[12];

    float* out = (float*)d_out;

    float* qb    = (float*)d_ws;                       // M*D f32
    float* kb    = qb + (size_t)M_DIM * D_DIM;         // M*D f32
    float* vbuf  = kb + (size_t)M_DIM * D_DIM;         // M*D f32
    float* bpart = vbuf + (size_t)M_DIM * D_DIM;       // M*16 f32
    float* beta  = bpart + (size_t)M_DIM * 16;         // M f32

    dim3 g1(M_DIM / BM, D_DIM / BN, 4);                // 128 x 16 x 4
    proj_gemm<<<g1, 128, 0, stream>>>(x, Wq, bq, Wk, bk, Wv, bv, Wb1, bb1, Wb2,
                                      qb, kb, vbuf, bpart);
    beta_finalize<<<M_DIM / 256, 256, 0, stream>>>(bpart, bb2, beta);
    delta_scan<<<B_DIM * 32, 256, 0, stream>>>(qb, kb, vbuf, beta, out);
    layernorm_inplace<<<M_DIM, 256, 0, stream>>>(out, lnw, lnb);
}
